// CartesianCSWinBlock_46024869544414
// MI455X (gfx1250) — compile-verified, matrix operand order checked by reading, not run on hardware
//
#include <hip/hip_runtime.h>
#include <hip/hip_bf16.h>

// ---------------------------------------------------------------------------
// CDNA5 (gfx1250) CSWin block. All GEMMs on v_wmma_f32_16x16x32_bf16.
// WMMA operands are fragment-contiguous in LDS (2x ds_load_b128 per operand).
// Tile staging uses GLOBAL_LOAD_ASYNC_TO_LDS_B128 when the toolchain exposes
// the builtin (ASYNCcnt-tracked), otherwise falls back to uint4 copies.
// ---------------------------------------------------------------------------

typedef __attribute__((ext_vector_type(16))) __bf16 bf16x16;
typedef __attribute__((ext_vector_type(8)))  __bf16 bf16x8;
typedef __attribute__((ext_vector_type(8)))  float  floatx8;
typedef int v4i __attribute__((vector_size(16)));
typedef __attribute__((address_space(1))) v4i* as1_v4i_ptr;   // global int4*
typedef __attribute__((address_space(3))) v4i* as3_v4i_ptr;   // LDS int4*

#define DEVINL __device__ __forceinline__

#if defined(__AMDGCN__)
#if __has_builtin(__builtin_amdgcn_global_load_async_to_lds_b128) && \
    __has_builtin(__builtin_amdgcn_s_wait_asynccnt)
#define CSWIN_ASYNC 1
#endif
#endif

// 16-byte global -> LDS stage (async when available)
DEVINL void stage16(void* lds, const void* glob) {
#ifdef CSWIN_ASYNC
    __builtin_amdgcn_global_load_async_to_lds_b128(
        (as1_v4i_ptr)glob, (as3_v4i_ptr)lds, 0, 0);
#else
    *(uint4*)lds = *(const uint4*)glob;
#endif
}

DEVINL void stage_fence() {
#ifdef CSWIN_ASYNC
    __builtin_amdgcn_s_wait_asynccnt(0);
#endif
}

DEVINL void stage_u4(void* dst, const void* src, int n16, int tid, int nthr) {
    uint4* d = (uint4*)dst;
    const uint4* s = (const uint4*)src;
    for (int i = tid; i < n16; i += nthr) stage16(d + i, s + i);
}

// ISA 7.12.2 K-swizzle for 16-bit fragments:
// element e of lane L holds K = ((e<8)?e:e+8) + 8*(L>>4)
// Inverse (packing): for K-in-tile kk: lane-half=(kk>>3)&1 ; e=(kk&7)+((kk>>4)<<3)

DEVINL floatx8 zero8() {
    floatx8 z;
#pragma unroll
    for (int i = 0; i < 8; ++i) z[i] = 0.f;
    return z;
}

DEVINL floatx8 wmma_bf16(bf16x16 a, bf16x16 b, floatx8 c) {
    return __builtin_amdgcn_wmma_f32_16x16x32_bf16(false, a, false, b,
                                                   (short)0, c, false, false);
}

// A fragment from row-major [rows][ld] buffer: per lane 8+8 contiguous halves.
DEVINL bf16x16 load_a_vec(const __bf16* base, int ld, int row0, int k0, int lane) {
    const __bf16* p = base + (row0 + (lane & 15)) * ld + k0 + ((lane >> 4) << 3);
    bf16x8 lo = *(const bf16x8*)(p);        // K = k0+8h .. +7
    bf16x8 hi = *(const bf16x8*)(p + 16);   // K = k0+16+8h .. +7
    return __builtin_shufflevector(lo, hi, 0, 1, 2, 3, 4, 5, 6, 7,
                                   8, 9, 10, 11, 12, 13, 14, 15);
}

// Fragment-packed tile ([32 lanes][16 elems], 1KB): one 32B vector per lane.
DEVINL bf16x16 load_packed(const __bf16* tile, int lane) {
    return *(const bf16x16*)(tile + lane * 16);
}

// ---------------------------------------------------------------------------
// Kernel: pack fp32 row-major weight [K][N] into bf16 B-fragment order:
// dst[ ((nt*Ktiles + kt)*32 + lane)*16 + e ]
// ---------------------------------------------------------------------------
__global__ void packB_kernel(const float* __restrict__ src, __bf16* __restrict__ dst,
                             int K, int N) {
    int i = blockIdx.x * 256 + threadIdx.x;
    if (i >= K * N) return;
    int k = i / N, n = i % N;
    int kt = k >> 5, kk = k & 31, nt = n >> 4, nn = n & 15;
    int lane = nn + (((kk >> 3) & 1) << 4);
    int e = (kk & 7) + ((kk >> 4) << 3);
    int Ktiles = K >> 5;
    dst[(((size_t)nt * Ktiles + kt) * 32 + lane) * 16 + e] = (__bf16)src[i];
}

// ---------------------------------------------------------------------------
// Kernel: LayerNorm1. x NCHW -> xnc channel-major bf16 [128][T].
// ---------------------------------------------------------------------------
__global__ __launch_bounds__(256) void ln1_kernel(
    const float* __restrict__ x, const float* __restrict__ g,
    const float* __restrict__ bb, __bf16* __restrict__ xnc, int T, int HW) {
    int t = blockIdx.x * 256 + threadIdx.x;
    int b = t / HW, pos = t % HW;
    const float* xb = x + ((size_t)b * 128) * HW + pos;
    float sum = 0.f, sq = 0.f;
    for (int c = 0; c < 128; ++c) {
        float v = xb[(size_t)c * HW];
        sum += v; sq += v * v;
    }
    float mean = sum * (1.f / 128.f);
    float var  = sq * (1.f / 128.f) - mean * mean;
    float inv  = rsqrtf(var + 1e-5f);
    for (int c = 0; c < 128; ++c) {
        float v = xb[(size_t)c * HW];
        xnc[(size_t)c * T + t] = (__bf16)((v - mean) * inv * g[c] + bb[c]);
    }
}

// ---------------------------------------------------------------------------
// Kernel: one CSWin window (64 tok x 64 ch) per 4-wave block; wave == head.
// QKV epilogue scatters Q/K/V into fragment-ready LDS so the QK^T and PV
// inner loops read contiguous fragments.
// ---------------------------------------------------------------------------
template <int Wh, int Ww, int CO>
__global__ __launch_bounds__(128) void attn_kernel(
    const __bf16* __restrict__ xnc, const __bf16* __restrict__ wqkv_p,
    const __bf16* __restrict__ wproj_p, const float* __restrict__ bproj,
    const float* __restrict__ table, float* __restrict__ ybr,
    int T, int H, int W) {
    extern __shared__ char smem[];
    __bf16* sX  = (__bf16*)(smem);             // 64x64 row-major      (8KB)
    __bf16* sW  = (__bf16*)(smem + 8192);      // packed qkv W, then proj (24KB)
    __bf16* sQ  = (__bf16*)(smem + 32768);     // [4h][4mt][32][16] A-frag, K-pad (16KB)
    __bf16* sK  = (__bf16*)(smem + 49152);     // [4h][4nt][32][16] B-frag, K-pad (16KB)
    __bf16* sV  = (__bf16*)(smem + 65536);     // [4h][2kt][32][16] B-frag (8KB)
    __bf16* sPt = (__bf16*)(smem + 73728);     // [4h][16][64] probs staging (8KB)
    __bf16* sO  = (__bf16*)(smem + 81920);     // 64x64 row-major AV out (8KB)

    const int tid = threadIdx.x, lane = tid & 31, wave = tid >> 5;
    const int HW = H * W;
    const int nWw = W / Ww;
    const int winPerB = (H / Wh) * nWw;
    const int wid = blockIdx.x;
    const int b  = wid / winPerB;
    const int r  = wid % winPerB;
    const int h0 = (r / nWw) * Wh, w0 = (r % nWw) * Ww;

    auto tok = [&](int n) {
        int i = n / Ww, j = n % Ww;
        return b * HW + (h0 + i) * W + (w0 + j);
    };

    // packed qkv weight (async linear copy)
    stage_u4(sW, wqkv_p, 12288 / 8, tid, 128);
    // zero K-padded regions of sQ/sK (32KB contiguous span)
    {
        int* z = (int*)(smem + 32768);
        for (int i = tid; i < 8192; i += 128) z[i] = 0;
    }
    // stage window activations (gather from channel-major xnc)
    for (int idx = tid; idx < 64 * 64; idx += 128) {
        int n = idx >> 6, c = idx & 63;
        sX[idx] = xnc[(size_t)(CO + c) * T + tok(n)];
    }
    stage_fence();
    __syncthreads();

    // ---- QKV: [64x64] @ [64x192]; scatter results into frag-ready buffers
    for (int tile = wave; tile < 48; tile += 4) {
        int mt = tile & 3, nt = tile >> 2;
        floatx8 acc = zero8();
#pragma unroll
        for (int kt = 0; kt < 2; ++kt) {
            bf16x16 a  = load_a_vec(sX, 64, mt * 16, kt * 32, lane);
            bf16x16 bm = load_packed(sW + (nt * 2 + kt) * 512, lane);
            acc = wmma_bf16(a, bm, acc);
        }
        int dd = lane & 15;
#pragma unroll
        for (int v = 0; v < 8; ++v) {
            int m = mt * 16 + v + ((lane >> 4) << 3);
            if (nt < 4) {            // Q -> A-frag layout, pre-scaled 1/sqrt(16)
                int hd = nt;
                sQ[(((hd * 4 + mt) * 32) + (m & 15) + ((dd >> 3) << 4)) * 16 + (dd & 7)] =
                    (__bf16)(acc[v] * 0.25f);
            } else if (nt < 8) {     // K -> B-frag layout (K^T: k=dd, n=token m)
                int hd = nt - 4;
                sK[(((hd * 4 + (m >> 4)) * 32) + (m & 15) + ((dd >> 3) << 4)) * 16 + (dd & 7)] =
                    (__bf16)acc[v];
            } else {                 // V -> B-frag layout (k=token m, n=dd)
                int hd = nt - 8;
                int kk = m & 31;
                sV[(((hd * 2 + (m >> 5)) * 32) + dd + (((kk >> 3) & 1) << 4)) * 16 +
                   (kk & 7) + ((kk >> 4) << 3)] = (__bf16)acc[v];
            }
        }
    }
    __syncthreads();
    // overlay packed proj weight in sW (consumed after the next barrier)
    stage_u4(sW, wproj_p, 4096 / 8, tid, 128);

    // ---- per-head attention; wave == head
    const int hd = wave;
    __bf16* sPh = sPt + hd * 16 * 64;
    for (int mt = 0; mt < 4; ++mt) {
        floatx8 s[4];
#pragma unroll
        for (int nt = 0; nt < 4; ++nt) {
            bf16x16 a  = load_packed(sQ + (hd * 4 + mt) * 512, lane);
            bf16x16 bm = load_packed(sK + (hd * 4 + nt) * 512, lane);
            s[nt] = wmma_bf16(a, bm, zero8());
        }
        // relative-position bias + row softmax (rows live in 16-lane halves)
#pragma unroll
        for (int v = 0; v < 8; ++v) {
            int q  = mt * 16 + v + ((lane >> 4) << 3);
            int iq = q / Ww, jq = q % Ww;
            float mx = -3.4e38f;
#pragma unroll
            for (int nt = 0; nt < 4; ++nt) {
                int key = nt * 16 + (lane & 15);
                int ik = key / Ww, jk = key % Ww;
                int ridx = (iq - ik + Wh - 1) * (2 * Ww - 1) + (jq - jk + Ww - 1);
                s[nt][v] = s[nt][v] + table[ridx * 4 + hd];
                mx = fmaxf(mx, s[nt][v]);
            }
#pragma unroll
            for (int msk = 8; msk >= 1; msk >>= 1) mx = fmaxf(mx, __shfl_xor(mx, msk, 32));
            float sum = 0.f;
#pragma unroll
            for (int nt = 0; nt < 4; ++nt) {
                float e = __expf(s[nt][v] - mx);
                s[nt][v] = e; sum += e;
            }
#pragma unroll
            for (int msk = 8; msk >= 1; msk >>= 1) sum += __shfl_xor(sum, msk, 32);
            float invs = 1.f / sum;
            int qr = v + ((lane >> 4) << 3);
#pragma unroll
            for (int nt = 0; nt < 4; ++nt)
                sPh[qr * 64 + nt * 16 + (lane & 15)] = (__bf16)(s[nt][v] * invs);
        }
        // ---- O[16x16] = P[16x64] @ V[64x16]
        floatx8 acc = zero8();
#pragma unroll
        for (int kt = 0; kt < 2; ++kt) {
            bf16x16 a  = load_a_vec(sPh, 64, 0, kt * 32, lane);
            bf16x16 bm = load_packed(sV + (hd * 2 + kt) * 512, lane);
            acc = wmma_bf16(a, bm, acc);
        }
        int dd = lane & 15;
#pragma unroll
        for (int v = 0; v < 8; ++v) {
            int m = mt * 16 + v + ((lane >> 4) << 3);
            sO[m * 64 + hd * 16 + dd] = (__bf16)acc[v];
        }
    }
    stage_fence();            // proj weight staged above must be in LDS
    __syncthreads();

    // ---- proj: [64x64] @ [64x64] + b_proj + xn residual -> branch buffer
    for (int tile = wave; tile < 16; tile += 4) {
        int mt = tile >> 2, nt = tile & 3;
        floatx8 acc = zero8();
#pragma unroll
        for (int kt = 0; kt < 2; ++kt) {
            bf16x16 a  = load_a_vec(sO, 64, mt * 16, kt * 32, lane);
            bf16x16 bm = load_packed(sW + (nt * 2 + kt) * 512, lane);
            acc = wmma_bf16(a, bm, acc);
        }
        int col = nt * 16 + (lane & 15);
        float bp = bproj[col];
#pragma unroll
        for (int v = 0; v < 8; ++v) {
            int m = mt * 16 + v + ((lane >> 4) << 3);
            float val = acc[v] + bp + (float)sX[m * 64 + col];
            ybr[(size_t)tok(m) * 128 + CO + col] = val;
        }
    }
}

// ---------------------------------------------------------------------------
// Kernel: y = x + branch_out ; LN2 -> bf16. One wave per token.
// ---------------------------------------------------------------------------
__global__ __launch_bounds__(256) void res_ln2_kernel(
    const float* __restrict__ x, const float* __restrict__ ybr,
    const float* __restrict__ g, const float* __restrict__ bb,
    float* __restrict__ ybuf, __bf16* __restrict__ hin, int T, int HW) {
    int wave = threadIdx.x >> 5, lane = threadIdx.x & 31;
    int t = blockIdx.x * 8 + wave;
    int b = t / HW, pos = t % HW;
    int c0 = lane * 4;
    const float4 br = reinterpret_cast<const float4*>(ybr + (size_t)t * 128)[lane];
    float brv[4] = {br.x, br.y, br.z, br.w};
    float v[4];
    float sum = 0.f, sq = 0.f;
#pragma unroll
    for (int u = 0; u < 4; ++u) {
        int c = c0 + u;
        float xv = x[((size_t)b * 128 + c) * HW + pos];
        v[u] = xv + brv[u];
        sum += v[u]; sq += v[u] * v[u];
    }
    float4 yv = {v[0], v[1], v[2], v[3]};
    reinterpret_cast<float4*>(ybuf + (size_t)t * 128)[lane] = yv;
#pragma unroll
    for (int msk = 16; msk >= 1; msk >>= 1) {
        sum += __shfl_xor(sum, msk, 32);
        sq  += __shfl_xor(sq,  msk, 32);
    }
    float mean = sum * (1.f / 128.f);
    float var  = sq * (1.f / 128.f) - mean * mean;
    float inv  = rsqrtf(var + 1e-5f);
#pragma unroll
    for (int u = 0; u < 4; ++u) {
        int c = c0 + u;
        hin[(size_t)t * 128 + c] = (__bf16)((v[u] - mean) * inv * g[c] + bb[c]);
    }
}

// ---------------------------------------------------------------------------
// Kernel: fc1 (128 -> 512) + exact-erf GELU, bf16 out. 64x64 tile per block.
// w1 pre-packed [nt][kt][32][16].
// ---------------------------------------------------------------------------
__global__ __launch_bounds__(128) void fc1_kernel(
    const __bf16* __restrict__ hin, const __bf16* __restrict__ w1p,
    const float* __restrict__ b1, __bf16* __restrict__ hbuf, int T) {
    extern __shared__ char smem[];
    __bf16* sA   = (__bf16*)(smem);           // 64x128 row-major (16KB)
    __bf16* sB   = (__bf16*)(smem + 16384);   // 16 packed tiles [nt][kt] (16KB)
    __bf16* sOut = (__bf16*)(smem + 32768);   // 64x64 (8KB)
    const int tid = threadIdx.x, lane = tid & 31, wave = tid >> 5;
    const int t0 = blockIdx.x * 64, c0 = blockIdx.y * 64;
    {   // A rows: 128 halves = 16 uint4 per row
        uint4* d = (uint4*)sA;
        for (int i = tid; i < 64 * 16; i += 128) {
            int row = i >> 4, u = i & 15;
            stage16(d + i, ((const uint4*)(hin + (size_t)(t0 + row) * 128)) + u);
        }
    }
    // B: packed tiles for nt = c0/16 .. +3, all kt -> contiguous span
    stage_u4(sB, w1p + (size_t)(c0 >> 4) * 4 * 512, 8192 / 8, tid, 128);
    __builtin_prefetch(hbuf + (size_t)t0 * 512 + c0, 1, 0);
    stage_fence();
    __syncthreads();
    const int mt = wave;
    floatx8 acc[4];
#pragma unroll
    for (int nt = 0; nt < 4; ++nt) acc[nt] = zero8();
#pragma unroll
    for (int kt = 0; kt < 4; ++kt) {
        bf16x16 a = load_a_vec(sA, 128, mt * 16, kt * 32, lane);
#pragma unroll
        for (int nt = 0; nt < 4; ++nt) {
            bf16x16 bm = load_packed(sB + (nt * 4 + kt) * 512, lane);
            acc[nt] = wmma_bf16(a, bm, acc[nt]);
        }
    }
#pragma unroll
    for (int nt = 0; nt < 4; ++nt) {
        int col = nt * 16 + (lane & 15);
        float bias = b1[c0 + col];
#pragma unroll
        for (int v = 0; v < 8; ++v) {
            int m = mt * 16 + v + ((lane >> 4) << 3);
            float xv = acc[nt][v] + bias;
            float gl = 0.5f * xv * (1.f + erff(xv * 0.70710678118654752f));
            sOut[m * 64 + col] = (__bf16)gl;
        }
    }
    __syncthreads();
    {   // 64 halves = 8 uint4 per row
        const uint4* s = (const uint4*)sOut;
        for (int i = tid; i < 64 * 8; i += 128) {
            int row = i >> 3, u = i & 7;
            ((uint4*)(hbuf + (size_t)(t0 + row) * 512 + c0))[u] = s[i];
        }
    }
}

// ---------------------------------------------------------------------------
// Kernel: fc2 (512 -> 128) + residual + NCHW transpose store.
// w2 pre-packed [nt][kt][32][16] (Ktiles = 16).
// ---------------------------------------------------------------------------
__global__ __launch_bounds__(128) void fc2_kernel(
    const __bf16* __restrict__ hbuf, const __bf16* __restrict__ w2p,
    const float* __restrict__ b2, const float* __restrict__ ybuf,
    float* __restrict__ out, int T, int HW) {
    extern __shared__ char smem[];
    __bf16* sA   = (__bf16*)(smem);                 // 64x128 (16KB) per K-chunk
    __bf16* sB   = (__bf16*)(smem + 16384);         // 32 packed tiles (32KB)
    float*  sOut = (float*)(smem + 16384 + 32768);  // 64x128 f32 (32KB)
    const int tid = threadIdx.x, lane = tid & 31, wave = tid >> 5;
    const int t0 = blockIdx.x * 64;
    const int mt = wave;
    floatx8 acc[8];
#pragma unroll
    for (int nt = 0; nt < 8; ++nt) acc[nt] = zero8();
    for (int kc = 0; kc < 4; ++kc) {
        {   // A rows: 128 halves = 16 uint4 per row (from 512-wide hbuf)
            uint4* d = (uint4*)sA;
            for (int i = tid; i < 64 * 16; i += 128) {
                int row = i >> 4, u = i & 15;
                stage16(d + i,
                        ((const uint4*)(hbuf + (size_t)(t0 + row) * 512 + kc * 128)) + u);
            }
        }
        {   // B: tiles [ntl 0..7][ktl 0..3] from packed [nt][16 kt]
            uint4* d = (uint4*)sB;
            for (int i = tid; i < 2048; i += 128) {
                int ntl = i >> 8, rem = i & 255;
                stage16(d + i,
                        ((const uint4*)(w2p + ((size_t)ntl * 16 + kc * 4) * 512)) + rem);
            }
        }
        if (kc < 3)
            __builtin_prefetch(hbuf + (size_t)(t0 + (tid & 63)) * 512 + (kc + 1) * 128, 0, 0);
        stage_fence();
        __syncthreads();
#pragma unroll
        for (int kt = 0; kt < 4; ++kt) {
            bf16x16 a = load_a_vec(sA, 128, mt * 16, kt * 32, lane);
#pragma unroll
            for (int nt = 0; nt < 8; ++nt) {
                bf16x16 bm = load_packed(sB + (nt * 4 + kt) * 512, lane);
                acc[nt] = wmma_bf16(a, bm, acc[nt]);
            }
        }
        __syncthreads();
    }
#pragma unroll
    for (int nt = 0; nt < 8; ++nt) {
        int col = nt * 16 + (lane & 15);
        float bias = b2[col];
#pragma unroll
        for (int v = 0; v < 8; ++v) {
            int m = mt * 16 + v + ((lane >> 4) << 3);
            float val = acc[nt][v] + bias + ybuf[(size_t)(t0 + m) * 128 + col];
            sOut[m * 128 + col] = val;
        }
    }
    __syncthreads();
    const int bidx = t0 / HW, pos0 = t0 % HW;   // HW % 64 == 0
    for (int idx = tid; idx < 64 * 128; idx += 128) {
        int c = idx >> 6, m = idx & 63;
        out[((size_t)bidx * 128 + c) * HW + pos0 + m] = sOut[m * 128 + c];
    }
}

// ---------------------------------------------------------------------------
// Host-side launch
// ---------------------------------------------------------------------------
extern "C" void kernel_launch(void* const* d_in, const int* in_sizes, int n_in,
                              void* d_out, int out_size, void* d_ws, size_t ws_size,
                              hipStream_t stream) {
    const float* x        = (const float*)d_in[0];
    const float* table_r  = (const float*)d_in[1];
    const float* w_qkv_r  = (const float*)d_in[2];
    const float* w_proj_r = (const float*)d_in[3];
    const float* b_proj_r = (const float*)d_in[4];
    const float* table_a  = (const float*)d_in[5];
    const float* w_qkv_a  = (const float*)d_in[6];
    const float* w_proj_a = (const float*)d_in[7];
    const float* b_proj_a = (const float*)d_in[8];
    const float* ln1_g    = (const float*)d_in[9];
    const float* ln1_b    = (const float*)d_in[10];
    const float* ln2_g    = (const float*)d_in[11];
    const float* ln2_b    = (const float*)d_in[12];
    const float* w_fc1    = (const float*)d_in[13];
    const float* b_fc1    = (const float*)d_in[14];
    const float* w_fc2    = (const float*)d_in[15];
    const float* b_fc2    = (const float*)d_in[16];
    float* out = (float*)d_out;
    (void)in_sizes; (void)n_in; (void)out_size; (void)ws_size;

    const int B = 4, H = 256, W = 256;
    const int HW = H * W;
    const int T  = B * HW;                  // 262144 tokens

    // workspace layout (bytes)
    char* ws = (char*)d_ws;
    __bf16* xnc  = (__bf16*)(ws);                        // [128][T] bf16   64MB
    float*  ybr  = (float*)(ws + 67108864ull);           // [T][128] f32   128MB
    float*  ybuf = (float*)(ws + 201326592ull);          // [T][128] f32   128MB
    __bf16* hin  = (__bf16*)(ws + 335544320ull);         // [T][128] bf16   64MB
    __bf16* hbuf = (__bf16*)(ws + 402653184ull);         // [T][512] bf16  256MB
    char*   wb   = ws + 671088640ull;                    // packed bf16 weights
    __bf16* pwqkv_r = (__bf16*)(wb);
    __bf16* pwproj_r= (__bf16*)(wb + 24576);
    __bf16* pwqkv_a = (__bf16*)(wb + 24576 + 8192);
    __bf16* pwproj_a= (__bf16*)(wb + 24576 + 8192 + 24576);
    __bf16* pw1     = (__bf16*)(wb + 24576 + 8192 + 24576 + 8192);
    __bf16* pw2     = (__bf16*)(wb + 24576 + 8192 + 24576 + 8192 + 131072);

    // 1) pack weights to B-fragment bf16 layout
    packB_kernel<<<(12288 + 255) / 256, 256, 0, stream>>>(w_qkv_r, pwqkv_r, 64, 192);
    packB_kernel<<<(4096  + 255) / 256, 256, 0, stream>>>(w_proj_r, pwproj_r, 64, 64);
    packB_kernel<<<(12288 + 255) / 256, 256, 0, stream>>>(w_qkv_a, pwqkv_a, 64, 192);
    packB_kernel<<<(4096  + 255) / 256, 256, 0, stream>>>(w_proj_a, pwproj_a, 64, 64);
    packB_kernel<<<(65536 + 255) / 256, 256, 0, stream>>>(w_fc1, pw1, 128, 512);
    packB_kernel<<<(65536 + 255) / 256, 256, 0, stream>>>(w_fc2, pw2, 512, 128);

    // 2) LayerNorm1 -> channel-major bf16
    ln1_kernel<<<T / 256, 256, 0, stream>>>(x, ln1_g, ln1_b, xnc, T, HW);

    // 3) window attention branches (4096 windows each)
    const int nWinR = B * (H / 16) * (W / 4);
    const int nWinA = B * (H / 4) * (W / 16);
    const size_t attnLds = 90112;   // 88KB
    attn_kernel<16, 4, 0><<<nWinR, 128, attnLds, stream>>>(
        xnc, pwqkv_r, pwproj_r, b_proj_r, table_r, ybr, T, H, W);
    attn_kernel<4, 16, 64><<<nWinA, 128, attnLds, stream>>>(
        xnc, pwqkv_a, pwproj_a, b_proj_a, table_a, ybr, T, H, W);

    // 4) residual + LayerNorm2
    res_ln2_kernel<<<T / 8, 256, 0, stream>>>(x, ybr, ln2_g, ln2_b, ybuf, hin, T, HW);

    // 5) MLP fc1 + GELU
    dim3 g1(T / 64, 8);
    fc1_kernel<<<g1, 128, 40960, stream>>>(hin, pw1, b_fc1, hbuf, T);

    // 6) MLP fc2 + residual + NCHW transposed store
    fc2_kernel<<<T / 64, 128, 81920, stream>>>(hbuf, pw2, b_fc2, ybuf, out, T, HW);
}
